// ComplexMamba3Layer_4037269258855
// MI455X (gfx1250) — compile-verified, matrix operand order checked by reading, not run on hardware
//
#include <hip/hip_runtime.h>

// ---------------- problem constants ----------------
#define BB     2
#define SS     4096
#define DM     1024
#define GG     8
#define DGRP   128
#define NST    64
#define MM     (BB * SS)          // 8192 tokens
#define NCHUNK 64
#define CHLEN  64                 // NCHUNK * CHLEN == SS

typedef __attribute__((ext_vector_type(16))) __bf16 v16bf;
typedef __attribute__((ext_vector_type(8)))  float  v8f;

// ---------------- bf16 helpers (bit-exact RNE, no __bf16 arithmetic needed) ----
static __device__ __forceinline__ __bf16 f2bf(float f) {
    union { float f; unsigned u; } in; in.f = f;
    unsigned lsb = (in.u >> 16) & 1u;
    unsigned r = (in.u + 0x7FFFu + lsb) >> 16;
    union { unsigned short s; __bf16 b; } out; out.s = (unsigned short)r;
    return out.b;
}
static __device__ __forceinline__ float bf2f(__bf16 b) {
    union { unsigned short s; __bf16 b; } in; in.b = b;
    union { float f; unsigned u; } out; out.u = ((unsigned)in.s) << 16;
    return out.f;
}

union Frag128 { uint4 q[2]; v16bf v; };

// =================================================================================
// Batched bf16 WMMA GEMM: C[m,n] = sum_k A[m,k] * B[n,k]
//  - block tile 256(M) x 64(N); 8 waves in 4x2 grid; wave tile 64x32 (4x2 WMMA tiles)
//  - A/B K-slabs staged in LDS via global_load_async_to_lds_b128 (ASYNCcnt)
// mode 0: gate     : outF[m*ldc+n]                 = sigmoid(acc + aux0[n])
// mode 1: Bx       : outF[z*sC + m*ldc + n]        = acc * aux0[m*8+z]   (dt_mag)
// mode 2: C-proj   : outH[z*sC + m*ldc + coff + n] = bf16(acc)
// mode 3: out-proj : outF[m*ldc + 2n + coff]       = aux1[m*N+n] + rs*aux0[n]*acc
// =================================================================================
__global__ __launch_bounds__(256) void wmma_gemm_kernel(
    const __bf16* __restrict__ A, int lda, long long sA,
    const __bf16* __restrict__ Bm, int ldb, long long sB,
    float* __restrict__ outF, __bf16* __restrict__ outH,
    long long sC, int ldc, int coff,
    int M_, int N_, int K_, int mode,
    const float* __restrict__ aux0, const float* __restrict__ aux1,
    const float* __restrict__ scalar0)
{
    __shared__ __align__(16) char smem[16384 + 4096];  // A slab (256x32 bf16) + B slab (64x32 bf16)

    const int z    = blockIdx.z;
    const int nbN  = N_ >> 6;
    const int bm   = (int)blockIdx.x / nbN;
    const int bn   = (int)blockIdx.x % nbN;
    const int tid  = threadIdx.x;
    const int wid  = tid >> 5, lane = tid & 31;
    const int mg   = wid >> 1, ng  = wid & 1;          // wave position in 4x2 grid
    const int half = lane >> 4, l16 = lane & 15;
    const int blockM = bm << 8, blockN = bn << 6;

    const char* Ag = (const char*)(A  + (long long)z * sA);
    const char* Bg = (const char*)(Bm + (long long)z * sB);
    const unsigned ldsBase = (unsigned)(uintptr_t)(&smem[0]); // low 32 bits = LDS offset

    v8f acc[4][2];
#pragma unroll
    for (int t = 0; t < 4; ++t)
#pragma unroll
        for (int u = 0; u < 2; ++u)
#pragma unroll
            for (int j = 0; j < 8; ++j) acc[t][u][j] = 0.0f;

    for (int k0 = 0; k0 < K_; k0 += 32) {
        // ---- stage A slab: 1024 x 16B chunks; chunk c -> row c>>2, 16B piece c&3 ----
#pragma unroll
        for (int i = 0; i < 4; ++i) {
            int c = tid + (i << 8);
            int row = c >> 2, kq = c & 3;
            unsigned long long ga = (unsigned long long)(uintptr_t)
                (Ag + ((long long)(blockM + row) * lda + k0) * 2 + kq * 16);
            unsigned lo = ldsBase + (unsigned)(c * 16);
            asm volatile("global_load_async_to_lds_b128 %0, %1, off"
                         :: "v"(lo), "v"(ga) : "memory");
        }
        // ---- stage B slab: 256 x 16B chunks ----
        {
            int row = tid >> 2, kq = tid & 3;
            unsigned long long gb = (unsigned long long)(uintptr_t)
                (Bg + ((long long)(blockN + row) * ldb + k0) * 2 + kq * 16);
            unsigned lo = ldsBase + 16384u + (unsigned)(tid * 16);
            asm volatile("global_load_async_to_lds_b128 %0, %1, off"
                         :: "v"(lo), "v"(gb) : "memory");
        }
        if (k0 + 32 < K_) {                 // global_prefetch_b8 next K-slab
            __builtin_prefetch(Ag + ((long long)(blockM + (tid >> 2)) * lda + k0 + 32) * 2, 0, 1);
        }
        asm volatile("s_wait_asynccnt 0" ::: "memory");
        __syncthreads();

        // ---- B fragments (shared across the wave's 4 M-tiles) ----
        v16bf bfr[2];
#pragma unroll
        for (int u = 0; u < 2; ++u) {
            int n = ng * 32 + u * 16 + l16;
            int boff = 16384 + n * 64 + half * 16;
            Frag128 fb;
            fb.q[0] = *(const uint4*)(smem + boff);
            fb.q[1] = *(const uint4*)(smem + boff + 32);
            bfr[u] = fb.v;
        }
        // ---- A fragments + 8 WMMAs ----
#pragma unroll
        for (int t = 0; t < 4; ++t) {
            int row = mg * 64 + t * 16 + l16;
            int aoff = row * 64 + half * 16;
            Frag128 fa;
            fa.q[0] = *(const uint4*)(smem + aoff);
            fa.q[1] = *(const uint4*)(smem + aoff + 32);
#pragma unroll
            for (int u = 0; u < 2; ++u) {
                acc[t][u] = __builtin_amdgcn_wmma_f32_16x16x32_bf16(
                    false, fa.v, false, bfr[u], (short)0, acc[t][u], false, false);
            }
        }
        __syncthreads();
    }

    const float rs = (mode == 3) ? scalar0[0] : 0.0f;
#pragma unroll
    for (int t = 0; t < 4; ++t)
#pragma unroll
        for (int u = 0; u < 2; ++u)
#pragma unroll
            for (int j = 0; j < 8; ++j) {
                int gm = blockM + mg * 64 + t * 16 + half * 8 + j;
                int gn = blockN + ng * 32 + u * 16 + l16;
                float v = acc[t][u][j];
                if (mode == 0) {
                    v = 1.0f / (1.0f + expf(-(v + aux0[gn])));
                    outF[(long long)gm * ldc + gn] = v;
                } else if (mode == 1) {
                    outF[(long long)z * sC + (long long)gm * ldc + gn] = v * aux0[gm * GG + z];
                } else if (mode == 2) {
                    outH[(long long)z * sC + (long long)gm * ldc + coff + gn] = f2bf(v);
                } else {
                    float o = aux1[(long long)gm * N_ + gn] + rs * aux0[gn] * v;
                    outF[(long long)gm * ldc + 2 * gn + coff] = o;
                }
            }
}

// ---------------- weight packing ----------------
__global__ void pack1_kernel(const float* __restrict__ in, __bf16* __restrict__ out,
                             long long count) {
    long long i = (long long)blockIdx.x * blockDim.x + threadIdx.x;
    if (i < count) out[i] = f2bf(in[i]);
}
// out[n, 0:Kh] = inA[n,:], out[n, Kh:2Kh] = sB * inB[n,:]
__global__ void pack2_kernel(const float* __restrict__ inA, const float* __restrict__ inB,
                             __bf16* __restrict__ out, int N_, int Kh, float sB) {
    long long i = (long long)blockIdx.x * blockDim.x + threadIdx.x;
    long long total = (long long)N_ * 2 * Kh;
    if (i >= total) return;
    int n = (int)(i / (2 * Kh));
    int k = (int)(i % (2 * Kh));
    float v = (k < Kh) ? inA[(long long)n * Kh + k] : sB * inB[(long long)n * Kh + (k - Kh)];
    out[i] = f2bf(v);
}

// ---------------- RMS norm -> xcat bf16 (M x 2048: [xr_n | xi_n]) --------------
__global__ __launch_bounds__(256) void rmsnorm_kernel(
    const float* __restrict__ xr, const float* __restrict__ xi,
    const float* __restrict__ nw, __bf16* __restrict__ xcat)
{
    __shared__ float red[256];
    const int m = blockIdx.x, t = threadIdx.x;
    float s = 0.0f;
    for (int i = t; i < DM; i += 256) {
        float a = xr[(long long)m * DM + i], b = xi[(long long)m * DM + i];
        s += a * a + b * b;
    }
    red[t] = s; __syncthreads();
    for (int off = 128; off > 0; off >>= 1) {
        if (t < off) red[t] += red[t + off];
        __syncthreads();
    }
    float inv = rsqrtf(red[0] / (float)DM + 1e-6f);
    for (int i = t; i < DM; i += 256) {
        float w = nw[i] * inv;
        xcat[(long long)m * 2048 + i]        = f2bf(xr[(long long)m * DM + i] * w);
        xcat[(long long)m * 2048 + 1024 + i] = f2bf(xi[(long long)m * DM + i] * w);
    }
}

// ---------------- selective rotation -------------------------------------------
__global__ void rotate_kernel(const __bf16* __restrict__ xcat, const float* __restrict__ g,
                              const float* __restrict__ theta,
                              float* __restrict__ xrr, float* __restrict__ xir,
                              long long total)
{
    long long idx = (long long)blockIdx.x * blockDim.x + threadIdx.x;
    if (idx >= total) return;
    long long m = idx >> 10;
    int d = (int)(idx & 1023);
    float a = bf2f(xcat[m * 2048 + d]);
    float b = bf2f(xcat[m * 2048 + 1024 + d]);
    float ang = theta[d >> 3] * g[idx];
    float c = cosf(ang), s = sinf(ang);
    xrr[idx] = a * c - b * s;
    xir[idx] = a * s + b * c;
}

// ---------------- depthwise causal conv + act gate + dt + A --------------------
__global__ __launch_bounds__(256) void conv_dt_A_kernel(
    const float* __restrict__ xrr, const float* __restrict__ xir,
    const float* __restrict__ conv_w, const float* __restrict__ conv_b,
    const float* __restrict__ act_thresh,
    const float* __restrict__ dt_w, const float* __restrict__ dt_b,
    const float* __restrict__ log_A_mag, const float* __restrict__ A_phase,
    __bf16* __restrict__ xg, float* __restrict__ dtmag,
    float* __restrict__ Aro, float* __restrict__ Aio)
{
    __shared__ float sdt[16];
    const int m = blockIdx.x, t = threadIdx.x;
    const int s = m & (SS - 1);
    if (t < 16) sdt[t] = 0.0f;
    __syncthreads();
    const float eth = expf(act_thresh[0]);
#pragma unroll
    for (int i = 0; i < 4; ++i) {
        int d = t + 256 * i;
        float cr = conv_b[2 * d], ci = conv_b[2 * d + 1];
#pragma unroll
        for (int j = 0; j < 4; ++j) {
            int sp = s - 3 + j;
            if (sp >= 0) {
                long long p = (long long)(m - 3 + j) * DM + d;
                cr += conv_w[d * 16 + j]      * xrr[p];   // o=2d,   ic=0
                ci += conv_w[d * 16 + 12 + j] * xir[p];   // o=2d+1, ic=1
            }
        }
        float gate = 1.0f - expf(-(cr * cr + ci * ci) * eth);
        cr *= gate; ci *= gate;
        int g = d >> 7, dl = d & 127;
        long long base = ((long long)g * MM + m) * 256;
        xg[base + dl]       = f2bf(cr);
        xg[base + 128 + dl] = f2bf(ci);
        atomicAdd(&sdt[2 * g],     cr * dt_w[dl]       + ci * dt_w[128 + dl]);        // ds_add_f32
        atomicAdd(&sdt[2 * g + 1], cr * dt_w[256 + dl] + ci * dt_w[256 + 128 + dl]);
    }
    __syncthreads();
    for (int i = t; i < GG * NST; i += 256) {
        int g = i >> 6, n = i & 63;
        float dm = expf(sdt[2 * g]     + dt_b[0]); dm = fminf(fmaxf(dm, 1e-4f), 2.0f);
        float dp = expf(sdt[2 * g + 1] + dt_b[1]); dp = fminf(fmaxf(dp, 1e-4f), 2.0f);
        if (n == 0) dtmag[m * GG + g] = dm;
        float nla = -logf(1.0f + expf(log_A_mag[g * NST + n]));   // -softplus
        float Am  = expf(dm * nla);
        float ang = dp * A_phase[g * NST + n];
        long long idx = ((long long)m * GG + g) * NST + n;
        Aro[idx] = Am * cosf(ang);
        Aio[idx] = Am * sinf(ang);
    }
}

// ---------------- 3-pass chunked complex scan ----------------------------------
__global__ __launch_bounds__(64) void scan_pass1(
    const float* __restrict__ Ar, const float* __restrict__ Ai,
    const float* __restrict__ Xr, const float* __restrict__ Xi,
    float* __restrict__ cAr, float* __restrict__ cAi,
    float* __restrict__ cHr, float* __restrict__ cHi)
{
    const int blk = blockIdx.x;               // bg*NCHUNK + chunk
    const int bg = blk / NCHUNK, chunk = blk % NCHUNK;
    const int b = bg >> 3, g = bg & 7, n = threadIdx.x;
    long long mbase = (long long)b * SS + (long long)chunk * CHLEN;
    float ar = 1.0f, ai = 0.0f, hr = 0.0f, hi = 0.0f;
    for (int stp = 0; stp < CHLEN; ++stp) {
        long long idx = ((mbase + stp) * GG + g) * NST + n;
        float tr = Ar[idx], ti = Ai[idx], xr = Xr[idx], xi = Xi[idx];
        float nar = tr * ar - ti * ai, nai = tr * ai + ti * ar;
        float nhr = tr * hr - ti * hi + xr, nhi = tr * hi + ti * hr + xi;
        ar = nar; ai = nai; hr = nhr; hi = nhi;
    }
    long long ci = (long long)blk * NST + n;
    cAr[ci] = ar; cAi[ci] = ai; cHr[ci] = hr; cHi[ci] = hi;
}

__global__ __launch_bounds__(64) void scan_pass2(
    const float* __restrict__ cAr, const float* __restrict__ cAi,
    const float* __restrict__ cHr, const float* __restrict__ cHi,
    float* __restrict__ inR, float* __restrict__ inI)
{
    const int bg = blockIdx.x, n = threadIdx.x;
    float hr = 0.0f, hi = 0.0f;
    for (int c = 0; c < NCHUNK; ++c) {
        long long ci = ((long long)bg * NCHUNK + c) * NST + n;
        inR[ci] = hr; inI[ci] = hi;
        float tr = cAr[ci], ti = cAi[ci];
        float nhr = tr * hr - ti * hi + cHr[ci];
        float nhi = tr * hi + ti * hr + cHi[ci];
        hr = nhr; hi = nhi;
    }
}

__global__ __launch_bounds__(64) void scan_pass3(
    const float* __restrict__ Ar, const float* __restrict__ Ai,
    const float* __restrict__ Xr, const float* __restrict__ Xi,
    const float* __restrict__ inR, const float* __restrict__ inI,
    __bf16* __restrict__ hcat)
{
    const int blk = blockIdx.x;
    const int bg = blk / NCHUNK, chunk = blk % NCHUNK;
    const int b = bg >> 3, g = bg & 7, n = threadIdx.x;
    long long mbase = (long long)b * SS + (long long)chunk * CHLEN;
    long long ci = (long long)blk * NST + n;
    float hr = inR[ci], hi = inI[ci];
    for (int stp = 0; stp < CHLEN; ++stp) {
        long long m = mbase + stp;
        long long idx = (m * GG + g) * NST + n;
        float tr = Ar[idx], ti = Ai[idx];
        float nhr = tr * hr - ti * hi + Xr[idx];
        float nhi = tr * hi + ti * hr + Xi[idx];
        hr = nhr; hi = nhi;
        long long hb = ((long long)g * MM + m) * 128;
        hcat[hb + n]      = f2bf(hr);
        hcat[hb + 64 + n] = f2bf(hi);
    }
}

// =================================================================================
extern "C" void kernel_launch(void* const* d_in, const int* in_sizes, int n_in,
                              void* d_out, int out_size, void* d_ws, size_t ws_size,
                              hipStream_t stream) {
    (void)in_sizes; (void)n_in; (void)out_size; (void)ws_size;
    const float* x_real     = (const float*)d_in[0];
    const float* x_imag     = (const float*)d_in[1];
    const float* norm_w     = (const float*)d_in[2];
    const float* sg_theta   = (const float*)d_in[3];
    const float* sg_wg      = (const float*)d_in[4];
    const float* sg_bg      = (const float*)d_in[5];
    const float* conv_w     = (const float*)d_in[6];
    const float* conv_b     = (const float*)d_in[7];
    const float* act_thresh = (const float*)d_in[8];
    const float* ssm_scale  = (const float*)d_in[9];
    const float* res_scale  = (const float*)d_in[10];
    const float* log_A_mag  = (const float*)d_in[11];
    const float* A_phase    = (const float*)d_in[12];
    const float* Bp_wr      = (const float*)d_in[13];
    const float* Bp_wi      = (const float*)d_in[14];
    const float* Cp_wr      = (const float*)d_in[15];
    const float* Cp_wi      = (const float*)d_in[16];
    const float* dt_w       = (const float*)d_in[17];
    const float* dt_b       = (const float*)d_in[18];
    const float* out_wr     = (const float*)d_in[19];
    const float* out_wi     = (const float*)d_in[20];

    // ---- workspace bump allocator (256B aligned) ----
    char* ws = (char*)d_ws;
    size_t off = 0;
    auto alloc = [&](size_t bytes) -> char* {
        char* p = ws + off;
        off = (off + bytes + 255) & ~(size_t)255;
        return p;
    };
    __bf16* xcat  = (__bf16*)alloc((size_t)MM * 2048 * 2);
    __bf16* WgP   = (__bf16*)alloc((size_t)DM * 2048 * 2);
    float*  gbuf  = (float*) alloc((size_t)MM * DM * 4);
    float*  xrr   = (float*) alloc((size_t)MM * DM * 4);
    float*  xir   = (float*) alloc((size_t)MM * DM * 4);
    __bf16* xg    = (__bf16*)alloc((size_t)GG * MM * 256 * 2);
    float*  dtmag = (float*) alloc((size_t)MM * GG * 4);
    float*  Ar    = (float*) alloc((size_t)MM * GG * NST * 4);
    float*  Ai    = (float*) alloc((size_t)MM * GG * NST * 4);
    float*  Bxr   = (float*) alloc((size_t)MM * GG * NST * 4);
    float*  Bxi   = (float*) alloc((size_t)MM * GG * NST * 4);
    __bf16* BwR   = (__bf16*)alloc((size_t)NST * 256 * 2);
    __bf16* BwI   = (__bf16*)alloc((size_t)NST * 256 * 2);
    float*  cAr   = (float*) alloc((size_t)1024 * NST * 4);
    float*  cAi   = (float*) alloc((size_t)1024 * NST * 4);
    float*  cHr   = (float*) alloc((size_t)1024 * NST * 4);
    float*  cHi   = (float*) alloc((size_t)1024 * NST * 4);
    float*  inR   = (float*) alloc((size_t)1024 * NST * 4);
    float*  inI   = (float*) alloc((size_t)1024 * NST * 4);
    __bf16* hcat  = (__bf16*)alloc((size_t)GG * MM * 128 * 2);
    __bf16* CwR   = (__bf16*)alloc((size_t)DGRP * 128 * 2);
    __bf16* CwI   = (__bf16*)alloc((size_t)DGRP * 128 * 2);
    __bf16* ycat  = (__bf16*)alloc((size_t)MM * 2048 * 2);
    __bf16* WoR   = (__bf16*)alloc((size_t)DM * 2048 * 2);
    __bf16* WoI   = (__bf16*)alloc((size_t)DM * 2048 * 2);

    // ---- 1. pack weights to bf16 ----
    {
        long long c = (long long)DM * 2048;
        pack1_kernel<<<(unsigned)((c + 255) / 256), 256, 0, stream>>>(sg_wg, WgP, c);
    }
    auto launch_pack2 = [&](const float* a, const float* b, __bf16* o, int N_, int Kh, float s) {
        long long c = (long long)N_ * 2 * Kh;
        pack2_kernel<<<(unsigned)((c + 255) / 256), 256, 0, stream>>>(a, b, o, N_, Kh, s);
    };
    launch_pack2(Bp_wr,  Bp_wi,  BwR, NST,  DGRP, -1.0f);
    launch_pack2(Bp_wi,  Bp_wr,  BwI, NST,  DGRP,  1.0f);
    launch_pack2(Cp_wr,  Cp_wi,  CwR, DGRP, NST,  -1.0f);
    launch_pack2(Cp_wi,  Cp_wr,  CwI, DGRP, NST,   1.0f);
    launch_pack2(out_wr, out_wi, WoR, DM,   DM,   -1.0f);
    launch_pack2(out_wi, out_wr, WoI, DM,   DM,    1.0f);

    // ---- 2. RMS norm ----
    rmsnorm_kernel<<<MM, 256, 0, stream>>>(x_real, x_imag, norm_w, xcat);

    // ---- 3. gate GEMM (WMMA, M=8192 N=1024 K=2048): blocks = (M/256)*(N/64) ----
    wmma_gemm_kernel<<<dim3((MM / 256) * (DM / 64), 1, 1), 256, 0, stream>>>(
        xcat, 2048, 0, WgP, 2048, 0, gbuf, nullptr, 0, DM, 0,
        MM, DM, 2048, 0, sg_bg, nullptr, nullptr);

    // ---- 4. rotation ----
    {
        long long total = (long long)MM * DM;
        rotate_kernel<<<(unsigned)((total + 255) / 256), 256, 0, stream>>>(
            xcat, gbuf, sg_theta, xrr, xir, total);
    }
    // ---- 5. conv + act gate + dt + A ----
    conv_dt_A_kernel<<<MM, 256, 0, stream>>>(
        xrr, xir, conv_w, conv_b, act_thresh, dt_w, dt_b,
        log_A_mag, A_phase, xg, dtmag, Ar, Ai);

    // ---- 6. Bx GEMMs (WMMA, batched over G: M=8192 N=64 K=256) ----
    wmma_gemm_kernel<<<dim3((MM / 256) * (NST / 64), 1, GG), 256, 0, stream>>>(
        xg, 256, (long long)MM * 256, BwR, 256, 0,
        Bxr, nullptr, NST, GG * NST, 0, MM, NST, 256, 1, dtmag, nullptr, nullptr);
    wmma_gemm_kernel<<<dim3((MM / 256) * (NST / 64), 1, GG), 256, 0, stream>>>(
        xg, 256, (long long)MM * 256, BwI, 256, 0,
        Bxi, nullptr, NST, GG * NST, 0, MM, NST, 256, 1, dtmag, nullptr, nullptr);

    // ---- 7. chunked complex scan ----
    scan_pass1<<<BB * GG * NCHUNK, NST, 0, stream>>>(Ar, Ai, Bxr, Bxi, cAr, cAi, cHr, cHi);
    scan_pass2<<<BB * GG, NST, 0, stream>>>(cAr, cAi, cHr, cHi, inR, inI);
    scan_pass3<<<BB * GG * NCHUNK, NST, 0, stream>>>(Ar, Ai, Bxr, Bxi, inR, inI, hcat);

    // ---- 8. C-proj GEMMs (WMMA, batched over G: M=8192 N=128 K=128) ----
    wmma_gemm_kernel<<<dim3((MM / 256) * (DGRP / 64), 1, GG), 256, 0, stream>>>(
        hcat, 128, (long long)MM * 128, CwR, 128, 0,
        nullptr, ycat, DGRP, 2048, 0, MM, DGRP, 128, 2, nullptr, nullptr, nullptr);
    wmma_gemm_kernel<<<dim3((MM / 256) * (DGRP / 64), 1, GG), 256, 0, stream>>>(
        hcat, 128, (long long)MM * 128, CwI, 128, 0,
        nullptr, ycat, DGRP, 2048, 1024, MM, DGRP, 128, 2, nullptr, nullptr, nullptr);

    // ---- 9. out-proj GEMMs + residual epilogue -> d_out (B,S,D,2) f32 ----
    {
        float* out = (float*)d_out;
        wmma_gemm_kernel<<<dim3((MM / 256) * (DM / 64), 1, 1), 256, 0, stream>>>(
            ycat, 2048, 0, WoR, 2048, 0, out, nullptr, 0, 2048, 0,
            MM, DM, 2048, 3, ssm_scale, x_real, res_scale);
        wmma_gemm_kernel<<<dim3((MM / 256) * (DM / 64), 1, 1), 256, 0, stream>>>(
            ycat, 2048, 0, WoI, 2048, 0, out, nullptr, 0, 2048, 1,
            MM, DM, 2048, 3, ssm_scale, x_imag, res_scale);
    }
}
